// MMRecBlock_82094004896185
// MI455X (gfx1250) — matrix operand order repaired and checked by reading, not compiled, not run on hardware
//
#include <hip/hip_runtime.h>
#include <cstdint>
#include <cstddef>

// ---------------------------------------------------------------------------
// MMRecBlock for MI455X (gfx1250).
//   B=4, S=512, D=1024, H=8, DH=128, INNER=256, FFN=4096, NSLOT=32
//
// Strategy: bf16 weights live in L2 (~30MB << 192MB L2).
//   Stage 1 (parallel): transpose weights to bf16 [N,K]; rmsnorm-stage xn;
//       hoisted WMMA GEMMs for q,k,v,z, hid, gamma over all 512 steps
//       (M = 512 step-tiles x 16 rows, rows 4..15 zero).
//   Stage 2 (sequential): persistent 64-block cooperative kernel; per step
//       P1 gate GEMM + slot attention; P2 ctx@Wo + h_t + mem update;
//       P3 residual + rmsnorm2; P4 FFN1(gelu); P5 FFN2 split-K x4; P6 reduce.
//   All GEMMs use v_wmma_f32_16x16x32_bf16 with f32 accumulation, with a
//   software-pipelined k-loop so fragment loads for iteration i+1 overlap the
//   WMMA of iteration i (partial s_wait_loadcnt instead of full drain).
// ---------------------------------------------------------------------------

#define B_     4
#define S_     512
#define D_     1024
#define H_     8
#define DH_    128
#define INNER_ 256
#define FFN_   4096
#define NSLOT_ 32
#define NBLK   64   // persistent kernel blocks (x 256 threads = 512 waves)

typedef __bf16 bf16;
typedef __attribute__((ext_vector_type(16))) __bf16 v16bf;
typedef __attribute__((ext_vector_type(8)))  float  v8f;

// ---------------- WMMA fragment loaders (CDNA5 wave32 layouts) -------------
// A 16x32 bf16: lanes 0-15 -> M=lane, K = {k0..k0+7} U {k0+16..k0+23}
//               lanes 16-31 -> M=lane-16, K = {k0+8..k0+15} U {k0+24..k0+31}
__device__ __forceinline__ v16bf load_a_frag(const bf16* A, int lda, int k0, int lane) {
  const int row = lane & 15;
  const int kh  = (lane >> 4) << 3;               // 0 or 8
  const bf16* p0 = A + (size_t)row * lda + k0 + kh;
  const bf16* p1 = p0 + 16;
  v16bf a;
#pragma unroll
  for (int i = 0; i < 8; ++i) { a[i] = p0[i]; a[i + 8] = p1[i]; }
  return a;
}

// B 32x16 bf16, weights stored transposed [N,K] row-major:
// lanes 0-15 -> N=lane, K={k0..k0+7} U {k0+16..k0+23} ; lanes 16-31 -> N=lane-16, K={k0+8..k0+15} U {k0+24..k0+31}
__device__ __forceinline__ v16bf load_b_frag(const bf16* Wt, int ldw, int colBase,
                                             int k0, int lane) {
  const int n  = lane & 15;
  const int kh = (lane >> 4) << 3;                // 0 or 8
  const bf16* p = Wt + (size_t)(colBase + n) * ldw + k0 + kh;
  v16bf b;
#pragma unroll
  for (int i = 0; i < 8; ++i) { b[i] = p[i]; b[i + 8] = p[i + 16]; }
  return b;
}

// Software-pipelined k-loop: issue loads for k0+32 before the WMMA of k0 so
// the in-order LOADcnt lets the compiler use partial waits and hide L2 latency
// behind the matrix op.
__device__ __forceinline__ v8f gemm_tile(const bf16* A, int lda, const bf16* Wt, int ldw,
                                         int colBase, int kbeg, int kend, int lane, v8f acc) {
  v16bf a0 = load_a_frag(A, lda, kbeg, lane);
  v16bf b0 = load_b_frag(Wt, ldw, colBase, kbeg, lane);
  for (int k0 = kbeg + 32; k0 < kend; k0 += 32) {
    v16bf a1 = load_a_frag(A, lda, k0, lane);
    v16bf b1 = load_b_frag(Wt, ldw, colBase, k0, lane);
    // stream the following B k-tile through L2 (global_prefetch_b8)
    __builtin_prefetch(Wt + (size_t)(colBase + (lane & 15)) * ldw + k0 + 32, 0, 1);
    acc = __builtin_amdgcn_wmma_f32_16x16x32_bf16(false, a0, false, b0, (short)0, acc,
                                                  false, false);
    a0 = a1; b0 = b1;
  }
  return __builtin_amdgcn_wmma_f32_16x16x32_bf16(false, a0, false, b0, (short)0, acc,
                                                 false, false);
}

__device__ __forceinline__ float sigmoidf_(float x) { return 1.f / (1.f + __expf(-x)); }
__device__ __forceinline__ float geluf_(float x) {
  return 0.5f * x * (1.f + erff(x * 0.7071067811865475f));
}

// ---------------------------- utility kernels ------------------------------
__global__ void zero_u32(uint32_t* p, size_t n) {
  for (size_t i = blockIdx.x * (size_t)blockDim.x + threadIdx.x; i < n;
       i += (size_t)gridDim.x * blockDim.x)
    p[i] = 0u;
}

__global__ void transpose_bf16(const float* __restrict__ W, bf16* __restrict__ Wt,
                               int K, int Nout) {
  const size_t total = (size_t)K * Nout;
  for (size_t idx = blockIdx.x * (size_t)blockDim.x + threadIdx.x; idx < total;
       idx += (size_t)gridDim.x * blockDim.x) {
    const int k = (int)(idx / Nout);
    const int n = (int)(idx - (size_t)k * Nout);
    Wt[(size_t)n * K + k] = (bf16)W[idx];
  }
}

// rmsnorm(x_t) for all t in parallel -> bf16 A-staging [S][16][D] (rows 0..3)
__global__ void rmsnorm_stage(const float* __restrict__ x, const float* __restrict__ w,
                              bf16* __restrict__ xnA) {
  const int t = blockIdx.x;
  __shared__ float red[256];
  for (int b = 0; b < B_; ++b) {
    const float* xt = x + ((size_t)b * S_ + t) * D_;
    float ss = 0.f;
    for (int d = threadIdx.x; d < D_; d += 256) { float v = xt[d]; ss += v * v; }
    red[threadIdx.x] = ss;
    __syncthreads();
    for (int o = 128; o; o >>= 1) {
      if (threadIdx.x < o) red[threadIdx.x] += red[threadIdx.x + o];
      __syncthreads();
    }
    const float nrm = sqrtf(red[0]) * (1.f / 32.f);      // ||x||/sqrt(D)
    const float inv = 1.f / (nrm + 1e-6f);
    for (int d = threadIdx.x; d < D_; d += 256)
      xnA[((size_t)t * 16 + b) * D_ + d] = (bf16)(w[d] * xt[d] * inv);
    __syncthreads();
  }
}

// ---------------- hoisted (parallel-over-t) WMMA GEMMs ---------------------
struct PreArgs {
  const bf16* A1; const bf16* W1; int K1;
  const bf16* A2; const bf16* W2; int K2;     // optional fused second GEMM
  const float* b1; const float* b2;
  float* outF;            // [S][4][Nout] fp32 (optional)
  bf16*  outB;            // [S][16][Nout] bf16 staging (optional)
  int Nout; int act;      // 0 none, 1 sigmoid, 2 tanh
};

__global__ void precompute_gemm(PreArgs p) {
  const int lane   = threadIdx.x & 31;
  const int wave   = (int)((blockIdx.x * blockDim.x + threadIdx.x) >> 5);
  const int nwaves = (int)((gridDim.x * blockDim.x) >> 5);
  const int coltiles = p.Nout >> 4;
  const int total = S_ * coltiles;
  for (int task = wave; task < total; task += nwaves) {
    const int t  = task / coltiles;
    const int ct = task - t * coltiles;
    v8f acc = {0.f, 0.f, 0.f, 0.f, 0.f, 0.f, 0.f, 0.f};
    acc = gemm_tile(p.A1 + (size_t)t * 16 * p.K1, p.K1, p.W1, p.K1,
                    ct * 16, 0, p.K1, lane, acc);
    if (p.A2)
      acc = gemm_tile(p.A2 + (size_t)t * 16 * p.K2, p.K2, p.W2, p.K2,
                      ct * 16, 0, p.K2, lane, acc);
    if (lane < 16) {
      const int col = ct * 16 + lane;
      const float bb = p.b1[col] + (p.b2 ? p.b2[col] : 0.f);
#pragma unroll
      for (int m = 0; m < B_; ++m) {                 // rows 0..3 = real batch
        float v = acc[m] + bb;
        if (p.act == 1) v = sigmoidf_(v);
        else if (p.act == 2) v = tanhf(v);
        if (p.outF) p.outF[((size_t)t * B_ + m) * p.Nout + col] = v;
        if (p.outB) p.outB[((size_t)t * 16 + m) * p.Nout + col] = (bf16)v;
      }
    }
  }
}

// ------------------------- persistent sequential scan ----------------------
__device__ __forceinline__ void grid_barrier(unsigned* bar, unsigned epoch) {
  __syncthreads();
  if (threadIdx.x == 0) {
    __threadfence();                               // release our stores
    atomicAdd(bar, 1u);
    const unsigned target = epoch * (unsigned)NBLK;
    while (*((volatile unsigned*)bar) < target) __builtin_amdgcn_s_sleep(2);
  }
  __syncthreads();
  __threadfence();                                 // acquire others' stores
}

struct SeqArgs {
  const float* x; float* out;
  const float *qbuf, *vbuf, *zbuf, *gbuf;          // [S][4][D] fp32 precomputed
  const bf16 *Wg, *Wo, *Wf1, *Wf2;                 // transposed bf16 weights
  const float *bg, *bo, *bf1, *bf2, *n2w;
  float *memk, *memv;                              // [4][32][D] working copies
  float *gate, *ctxf, *hprev, *hcur, *xres, *part; // small fp32 scratch
  bf16  *Ah, *Actx, *Axn2, *Affn;                  // [16,K] bf16 A staging
  unsigned* bar;
};

__global__ void seq_scan(SeqArgs s) {
  const int lane = threadIdx.x & 31;
  const int wave = (int)(blockIdx.x * (blockDim.x >> 5)) + (int)(threadIdx.x >> 5);
  unsigned epoch = 0;

  for (int t = 0; t < S_; ++t) {
    // -------- P1: gate = sigmoid(h_prev @ W_g + b_g)  +  slot attention ----
    if (wave < D_ / 16) {                                    // 64 tiles
      v8f acc = {0.f, 0.f, 0.f, 0.f, 0.f, 0.f, 0.f, 0.f};
      acc = gemm_tile(s.Ah, D_, s.Wg, D_, wave * 16, 0, D_, lane, acc);
      if (lane < 16) {
        const int col = wave * 16 + lane; const float bb = s.bg[col];
#pragma unroll
        for (int m = 0; m < B_; ++m) s.gate[m * D_ + col] = sigmoidf_(acc[m] + bb);
      }
    } else if (wave < 64 + B_ * H_) {                        // 32 (b,h) tasks
      const int task = wave - 64, b = task >> 3, h = task & 7;
      const float* q  = s.qbuf + ((size_t)t * B_ + b) * D_ + h * DH_;
      const float* kn = s.memk + ((size_t)b * NSLOT_ + lane) * D_ + h * DH_;
      float sc = 0.f;
      for (int d = 0; d < DH_; ++d) sc += q[d] * kn[d];      // lane = slot n
      sc *= 0.08838834764831845f;                            // 1/sqrt(128)
      float mx = sc;
#pragma unroll
      for (int o = 16; o; o >>= 1) mx = fmaxf(mx, __shfl_xor(mx, o));
      const float e = __expf(sc - mx);
      float ssum = e;
#pragma unroll
      for (int o = 16; o; o >>= 1) ssum += __shfl_xor(ssum, o);
      const float aw = e / ssum;
      float cx0 = 0.f, cx1 = 0.f, cx2 = 0.f, cx3 = 0.f;      // lane owns 4 dims
      for (int n = 0; n < NSLOT_; ++n) {
        const float an = __shfl(aw, n);
        const float* vn = s.memv + ((size_t)b * NSLOT_ + n) * D_ + h * DH_ + lane * 4;
        cx0 += an * vn[0]; cx1 += an * vn[1]; cx2 += an * vn[2]; cx3 += an * vn[3];
      }
      bf16* cb = s.Actx + (size_t)b * D_ + h * DH_ + lane * 4;
      cb[0] = (bf16)cx0; cb[1] = (bf16)cx1; cb[2] = (bf16)cx2; cb[3] = (bf16)cx3;
    }
    grid_barrier(s.bar, ++epoch);

    // -------- P2: attn_out = ctx @ Wo + bo   +   h_t + memory slot write ---
    if (wave < D_ / 16) {
      v8f acc = {0.f, 0.f, 0.f, 0.f, 0.f, 0.f, 0.f, 0.f};
      acc = gemm_tile(s.Actx, D_, s.Wo, D_, wave * 16, 0, D_, lane, acc);
      if (lane < 16) {
        const int col = wave * 16 + lane; const float bb = s.bo[col];
#pragma unroll
        for (int m = 0; m < B_; ++m) s.ctxf[m * D_ + col] = acc[m] + bb;
      }
    } else if (wave < 96) {                                  // h_t elementwise
      const int base = (wave - 64) * 128 + lane * 4;
      const int slot = t & (NSLOT_ - 1);
#pragma unroll
      for (int i = 0; i < 4; ++i) {
        const int e = base + i, b = e >> 10, col = e & (D_ - 1);
        const float z  = s.zbuf[(size_t)t * (B_ * D_) + e];
        const float ga = s.gbuf[(size_t)t * (B_ * D_) + e];
        const float hp = s.hprev[e];
        const float gt = s.gate[e];
        const float hnew = ga * hp + (1.f - ga) * z;
        const float ht = z * gt + ga * hp + 0.1f * hnew;
        s.hcur[e] = ht;
        s.Ah[e]   = (bf16)ht;                                // next-step gate A
        s.memk[((size_t)b * NSLOT_ + slot) * D_ + col] = ht;
        s.memv[((size_t)b * NSLOT_ + slot) * D_ + col] = ht;
      }
    }
    grid_barrier(s.bar, ++epoch);

    // -------- P3: x_res = x + h_t + attn + 0.1v ; rmsnorm2 -> Axn2 ---------
    if (wave < B_) {                                         // one wave per row
      const int b = wave;
      const float* xt = s.x + ((size_t)b * S_ + t) * D_;
      float loc[D_ / 32];
      float ss = 0.f;
#pragma unroll
      for (int j = 0; j < D_ / 32; ++j) {
        const int d = lane + j * 32;
        const float xr = xt[d] + s.hcur[b * D_ + d] + s.ctxf[b * D_ + d]
                       + 0.1f * s.vbuf[(size_t)t * (B_ * D_) + b * D_ + d];
        loc[j] = xr; ss += xr * xr;
      }
#pragma unroll
      for (int o = 16; o; o >>= 1) ss += __shfl_xor(ss, o);
      const float inv = 1.f / (sqrtf(ss) * (1.f / 32.f) + 1e-6f);
#pragma unroll
      for (int j = 0; j < D_ / 32; ++j) {
        const int d = lane + j * 32;
        s.xres[b * D_ + d]  = loc[j];
        s.hprev[b * D_ + d] = s.hcur[b * D_ + d];            // advance state
        s.Axn2[b * D_ + d]  = (bf16)(s.n2w[d] * loc[j] * inv);
      }
    }
    grid_barrier(s.bar, ++epoch);

    // -------- P4: FFN1 = gelu(xn2 @ Wf1 + b) -> bf16 staging ---------------
    if (wave < FFN_ / 16) {                                  // 256 tiles
      v8f acc = {0.f, 0.f, 0.f, 0.f, 0.f, 0.f, 0.f, 0.f};
      acc = gemm_tile(s.Axn2, D_, s.Wf1, D_, wave * 16, 0, D_, lane, acc);
      if (lane < 16) {
        const int col = wave * 16 + lane; const float bb = s.bf1[col];
#pragma unroll
        for (int m = 0; m < B_; ++m)
          s.Affn[m * FFN_ + col] = (bf16)geluf_(acc[m] + bb);
      }
    }
    grid_barrier(s.bar, ++epoch);

    // -------- P5: FFN2 split-K x4 partials ---------------------------------
    if (wave < 256) {
      const int ct = wave >> 2, kq = wave & 3;
      v8f acc = {0.f, 0.f, 0.f, 0.f, 0.f, 0.f, 0.f, 0.f};
      acc = gemm_tile(s.Affn, FFN_, s.Wf2, FFN_, ct * 16,
                      kq * 1024, (kq + 1) * 1024, lane, acc);
      if (lane < 16) {
        const int col = ct * 16 + lane;
#pragma unroll
        for (int m = 0; m < B_; ++m)
          s.part[(size_t)kq * (B_ * D_) + m * D_ + col] = acc[m];
      }
    }
    grid_barrier(s.bar, ++epoch);

    // -------- P6: out_t = x_res + sum(parts) + b_f2 ------------------------
    if (wave < 32) {
      const int base = wave * 128 + lane * 4;
#pragma unroll
      for (int i = 0; i < 4; ++i) {
        const int e = base + i, b = e >> 10, col = e & (D_ - 1);
        const float v = s.part[e] + s.part[B_ * D_ + e]
                      + s.part[2 * B_ * D_ + e] + s.part[3 * B_ * D_ + e];
        s.out[((size_t)b * S_ + t) * D_ + col] = s.xres[e] + v + s.bf2[col];
      }
    }
    grid_barrier(s.bar, ++epoch);
  }
}

// ------------------------------- host side ---------------------------------
extern "C" void kernel_launch(void* const* d_in, const int* in_sizes, int n_in,
                              void* d_out, int out_size, void* d_ws, size_t ws_size,
                              hipStream_t stream) {
  (void)in_sizes; (void)n_in; (void)out_size; (void)ws_size;
  const float* x     = (const float*)d_in[0];
  const float* mem_k = (const float*)d_in[1];
  const float* mem_v = (const float*)d_in[2];
  const float* W_q = (const float*)d_in[3],  *b_q = (const float*)d_in[4];
  const float* W_k = (const float*)d_in[5],  *b_k = (const float*)d_in[6];
  const float* W_v = (const float*)d_in[7],  *b_v = (const float*)d_in[8];
  const float* W_z = (const float*)d_in[9],  *b_z = (const float*)d_in[10];
  const float* W_g = (const float*)d_in[11], *b_g = (const float*)d_in[12];
  const float* W1m = (const float*)d_in[13], *b1m = (const float*)d_in[14];
  const float* Wcm = (const float*)d_in[15], *bcm = (const float*)d_in[16];
  const float* W2m = (const float*)d_in[17], *b2m = (const float*)d_in[18];
  const float* Wo  = (const float*)d_in[19], *bo  = (const float*)d_in[20];
  const float* Wf1 = (const float*)d_in[21], *bf1 = (const float*)d_in[22];
  const float* Wf2 = (const float*)d_in[23], *bf2 = (const float*)d_in[24];
  const float* n1w = (const float*)d_in[25];
  const float* n2w = (const float*)d_in[26];
  float* out = (float*)d_out;

  // ---- workspace carving (bump allocator, 256B aligned) ----
  char* ws = (char*)d_ws;
  size_t off = 0;
  auto alloc = [&](size_t bytes) -> void* {
    void* p = ws + off;
    off = (off + bytes + 255) & ~(size_t)255;
    return p;
  };

  // bf16 A-staging (must be zeroed each launch: rows 4..15 feed WMMA as zero)
  const size_t stag_beg = off;
  bf16* xnA  = (bf16*)alloc((size_t)S_ * 16 * D_ * 2);
  bf16* kA   = (bf16*)alloc((size_t)S_ * 16 * D_ * 2);
  bf16* zA   = (bf16*)alloc((size_t)S_ * 16 * D_ * 2);
  bf16* hidA = (bf16*)alloc((size_t)S_ * 16 * INNER_ * 2);
  bf16* Ah   = (bf16*)alloc((size_t)16 * D_ * 2);
  bf16* Actx = (bf16*)alloc((size_t)16 * D_ * 2);
  bf16* Axn2 = (bf16*)alloc((size_t)16 * D_ * 2);
  bf16* Affn = (bf16*)alloc((size_t)16 * FFN_ * 2);
  const size_t stag_end = off;

  // transposed bf16 weights [N,K] (fully overwritten each launch)
  bf16* Wqt  = (bf16*)alloc((size_t)D_ * D_ * 2);
  bf16* Wkt  = (bf16*)alloc((size_t)D_ * D_ * 2);
  bf16* Wvt  = (bf16*)alloc((size_t)D_ * D_ * 2);
  bf16* Wzt  = (bf16*)alloc((size_t)D_ * D_ * 2);
  bf16* Wgt  = (bf16*)alloc((size_t)D_ * D_ * 2);
  bf16* W1t  = (bf16*)alloc((size_t)INNER_ * D_ * 2);
  bf16* Wct  = (bf16*)alloc((size_t)INNER_ * D_ * 2);
  bf16* W2t  = (bf16*)alloc((size_t)D_ * INNER_ * 2);
  bf16* Wot  = (bf16*)alloc((size_t)D_ * D_ * 2);
  bf16* Wf1t = (bf16*)alloc((size_t)FFN_ * D_ * 2);
  bf16* Wf2t = (bf16*)alloc((size_t)D_ * FFN_ * 2);

  // fp32 precomputed activations [S][4][D]
  float* qbuf = (float*)alloc((size_t)S_ * B_ * D_ * 4);
  float* vbuf = (float*)alloc((size_t)S_ * B_ * D_ * 4);
  float* zbuf = (float*)alloc((size_t)S_ * B_ * D_ * 4);
  float* gbuf = (float*)alloc((size_t)S_ * B_ * D_ * 4);

  // sequential-kernel scratch
  float* memk = (float*)alloc((size_t)B_ * NSLOT_ * D_ * 4);
  float* memv = (float*)alloc((size_t)B_ * NSLOT_ * D_ * 4);
  float* gate = (float*)alloc((size_t)B_ * D_ * 4);
  float* ctxf = (float*)alloc((size_t)B_ * D_ * 4);
  float* hcur = (float*)alloc((size_t)B_ * D_ * 4);
  float* xres = (float*)alloc((size_t)B_ * D_ * 4);
  float* part = (float*)alloc((size_t)4 * B_ * D_ * 4);
  const size_t z2_beg = off;
  float* hprev = (float*)alloc((size_t)B_ * D_ * 4);   // must start at zero
  unsigned* bar = (unsigned*)alloc(256);               // barrier counter
  const size_t z2_end = off;

  // ---- init: zero staging + h0 + barrier, copy memory buffers ----
  zero_u32<<<8192, 256, 0, stream>>>((uint32_t*)(ws + stag_beg),
                                     (stag_end - stag_beg) / 4);
  zero_u32<<<64, 256, 0, stream>>>((uint32_t*)(ws + z2_beg), (z2_end - z2_beg) / 4);
  hipMemcpyAsync(memk, mem_k, (size_t)B_ * NSLOT_ * D_ * 4,
                 hipMemcpyDeviceToDevice, stream);
  hipMemcpyAsync(memv, mem_v, (size_t)B_ * NSLOT_ * D_ * 4,
                 hipMemcpyDeviceToDevice, stream);

  // ---- transpose + convert all weights to bf16 [N,K] ----
  transpose_bf16<<<8192, 256, 0, stream>>>(W_q, Wqt, D_, D_);
  transpose_bf16<<<8192, 256, 0, stream>>>(W_k, Wkt, D_, D_);
  transpose_bf16<<<8192, 256, 0, stream>>>(W_v, Wvt, D_, D_);
  transpose_bf16<<<8192, 256, 0, stream>>>(W_z, Wzt, D_, D_);
  transpose_bf16<<<8192, 256, 0, stream>>>(W_g, Wgt, D_, D_);
  transpose_bf16<<<2048, 256, 0, stream>>>(W1m, W1t, D_, INNER_);
  transpose_bf16<<<2048, 256, 0, stream>>>(Wcm, Wct, D_, INNER_);
  transpose_bf16<<<2048, 256, 0, stream>>>(W2m, W2t, INNER_, D_);
  transpose_bf16<<<8192, 256, 0, stream>>>(Wo,  Wot, D_, D_);
  transpose_bf16<<<8192, 256, 0, stream>>>(Wf1, Wf1t, D_, FFN_);
  transpose_bf16<<<8192, 256, 0, stream>>>(Wf2, Wf2t, FFN_, D_);

  // ---- stage xn = rmsnorm(x_t) for all t ----
  rmsnorm_stage<<<S_, 256, 0, stream>>>(x, n1w, xnA);

  // ---- hoisted WMMA GEMMs over all timesteps ----
  PreArgs pa{};
  pa.A1 = xnA; pa.K1 = D_; pa.A2 = nullptr; pa.K2 = 0; pa.b2 = nullptr;

  pa.W1 = Wqt; pa.b1 = b_q; pa.outF = qbuf; pa.outB = nullptr; pa.Nout = D_; pa.act = 0;
  precompute_gemm<<<2048, 256, 0, stream>>>(pa);
  pa.W1 = Wkt; pa.b1 = b_k; pa.outF = nullptr; pa.outB = kA;
  precompute_gemm<<<2048, 256, 0, stream>>>(pa);
  pa.W1 = Wvt; pa.b1 = b_v; pa.outF = vbuf; pa.outB = nullptr;
  precompute_gemm<<<2048, 256, 0, stream>>>(pa);
  pa.W1 = Wzt; pa.b1 = b_z; pa.outF = zbuf; pa.outB = zA;
  precompute_gemm<<<2048, 256, 0, stream>>>(pa);

  // hid = tanh(z@W1m + k@Wcm + b1m + bcm)
  PreArgs ph{};
  ph.A1 = zA; ph.W1 = W1t; ph.K1 = D_; ph.b1 = b1m;
  ph.A2 = kA; ph.W2 = Wct; ph.K2 = D_; ph.b2 = bcm;
  ph.outF = nullptr; ph.outB = hidA; ph.Nout = INNER_; ph.act = 2;
  precompute_gemm<<<2048, 256, 0, stream>>>(ph);

  // gamma = sigmoid(hid@W2m + b2m)
  PreArgs pg{};
  pg.A1 = hidA; pg.W1 = W2t; pg.K1 = INNER_; pg.b1 = b2m;
  pg.A2 = nullptr; pg.K2 = 0; pg.b2 = nullptr;
  pg.outF = gbuf; pg.outB = nullptr; pg.Nout = D_; pg.act = 1;
  precompute_gemm<<<2048, 256, 0, stream>>>(pg);

  // ---- persistent sequential scan ----
  SeqArgs sa{};
  sa.x = x; sa.out = out;
  sa.qbuf = qbuf; sa.vbuf = vbuf; sa.zbuf = zbuf; sa.gbuf = gbuf;
  sa.Wg = Wgt; sa.Wo = Wot; sa.Wf1 = Wf1t; sa.Wf2 = Wf2t;
  sa.bg = b_g; sa.bo = bo; sa.bf1 = bf1; sa.bf2 = bf2; sa.n2w = n2w;
  sa.memk = memk; sa.memv = memv;
  sa.gate = gate; sa.ctxf = ctxf; sa.hprev = hprev; sa.hcur = hcur;
  sa.xres = xres; sa.part = part;
  sa.Ah = Ah; sa.Actx = Actx; sa.Axn2 = Axn2; sa.Affn = Affn;
  sa.bar = bar;
  seq_scan<<<NBLK, 256, 0, stream>>>(sa);
}
